// SpatioTemporalGNN_56942676410677
// MI455X (gfx1250) — matrix operand order repaired and checked by reading, not compiled, not run on hardware
//
#include <hip/hip_runtime.h>
#include <hip/hip_bf16.h>

// SpatioTemporal GNN (GATv2 x2 per timestep -> LSTM -> MLP) for MI455X gfx1250.
// Dense node/LSTM GEMMs use v_wmma_f32_16x16x32_bf16 (wave32, one wave per
// 16x16 tile, weights pre-transposed for contiguous bf16 fragment loads).
// Edge phase (gather/leaky-relu/segment-softmax/scatter) stays f32 with
// wave-per-edge coalesced access + global f32 atomics (L2-resident tables).

#define TSEQ 6
#define FIN  64

typedef __attribute__((ext_vector_type(16))) __bf16 v16bf;
typedef __attribute__((ext_vector_type(8)))  __bf16 v8bf;
typedef __attribute__((ext_vector_type(8)))  float  v8f;

#define CDIV(a, b) (((a) + (b) - 1) / (b))

// ---------------------------------------------------------------- utilities
__device__ __forceinline__ float sigm(float x) { return 1.0f / (1.0f + __expf(-x)); }

// order-preserving float->uint key for atomicMax-based segment max
__device__ __forceinline__ unsigned fkey(float f) {
  unsigned u = __float_as_uint(f);
  return (u & 0x80000000u) ? ~u : (u | 0x80000000u);
}
__device__ __forceinline__ float funkey(unsigned k) {
  unsigned u = (k & 0x80000000u) ? (k ^ 0x80000000u) : ~k;
  return __uint_as_float(u);
}

__global__ void fill_f32(float* p, float v, long n) {
  long i = (long)blockIdx.x * blockDim.x + threadIdx.x;
  if (i < n) p[i] = v;
}
__global__ void fill_u32(unsigned* p, unsigned v, long n) {
  long i = (long)blockIdx.x * blockDim.x + threadIdx.x;
  if (i < n) p[i] = v;
}

// strided f32 -> bf16 row gather/convert (used for x[:,t,:], h, weights)
__global__ void cvt_rows_bf16(const float* __restrict__ in, __bf16* __restrict__ out,
                              long total, int rowlen, long instride, long inoff) {
  long i = (long)blockIdx.x * blockDim.x + threadIdx.x;
  if (i >= total) return;
  long r = i / rowlen;
  int  c = (int)(i - r * rowlen);
  out[i] = (__bf16)in[r * instride + inoff + c];
}

// W[K x Nout] f32 -> Wt[Nout x K] bf16
__global__ void transpose_cvt_bf16(const float* __restrict__ W, __bf16* __restrict__ Wt,
                                   int K, int Nout) {
  long i = (long)blockIdx.x * blockDim.x + threadIdx.x;
  if (i >= (long)K * Nout) return;
  int k = (int)(i / Nout), n = (int)(i - (long)k * Nout);
  Wt[(size_t)n * K + k] = (__bf16)W[i];
}

// mode 0: elu(y + b), mode 1: relu(y + b)
__global__ void bias_act(float* y, const float* __restrict__ b, long n, int rowlen, int mode) {
  long i = (long)blockIdx.x * blockDim.x + threadIdx.x;
  if (i >= n) return;
  float v = y[i] + b[i % rowlen];
  y[i] = (mode == 0) ? ((v > 0.f) ? v : (__expf(v) - 1.0f)) : fmaxf(v, 0.f);
}

// ------------------------------------------------------- WMMA bf16 GEMM
// Y[M x Nout] (+)= A[M x K](bf16) @ Wt[Nout x K]^T(bf16).  One wave per 16x16
// tile; K assumed multiple of 32, M/Nout multiples of 16.
__global__ __launch_bounds__(32)
void gemm_bf16(const __bf16* __restrict__ A, const __bf16* __restrict__ Wt,
               float* __restrict__ Y, int M, int K, int Nout, int acc) {
  const int lane = threadIdx.x & 31;
  const int m0 = blockIdx.x * 16;
  const int n0 = blockIdx.y * 16;
  const int hi = (lane >> 4) & 1;           // which half-wave
  const int rowA = m0 + (lane & 15);
  const int colB = n0 + (lane & 15);

  v8f c;
#pragma unroll
  for (int v = 0; v < 8; ++v)
    c[v] = acc ? Y[(size_t)(m0 + v + hi * 8) * Nout + colB] : 0.0f;

  const __bf16* arow = A + (size_t)rowA * K;
  const __bf16* brow = Wt + (size_t)colB * K;

  for (int k0 = 0; k0 < K; k0 += 32) {
    // A 16x32 bf16 layout: lanes 0-15 elems0-7 = K+0..7, elems8-15 = K+16..23;
    // lanes 16-31 shifted by +8 within each 16-wide half.
    v8bf alo = *(const v8bf*)(arow + k0 + hi * 8);
    v8bf ahi = *(const v8bf*)(arow + k0 + 16 + hi * 8);
    // B 32x16: lane holds col n0+(lane&15); same K layout as A.
    v8bf blo = *(const v8bf*)(brow + k0 + hi * 8);
    v8bf bhi = *(const v8bf*)(brow + k0 + 16 + hi * 8);
    v16bf av = __builtin_shufflevector(alo, ahi, 0,1,2,3,4,5,6,7,8,9,10,11,12,13,14,15);
    v16bf bv = __builtin_shufflevector(blo, bhi, 0,1,2,3,4,5,6,7,8,9,10,11,12,13,14,15);
    c = __builtin_amdgcn_wmma_f32_16x16x32_bf16(false, av, false, bv,
                                                (short)0, c, false, false);
  }

#pragma unroll
  for (int v = 0; v < 8; ++v)
    Y[(size_t)(m0 + v + hi * 8) * Nout + colB] = c[v];
}

// ------------------------------------------------------- GATv2 edge kernels
// one wave per edge; Hh heads of 64 channels each
__global__ void edge_logit(const float* __restrict__ xl, const float* __restrict__ xr,
                           const int* __restrict__ src, const int* __restrict__ dst,
                           const float* __restrict__ att, float* __restrict__ logit,
                           unsigned* __restrict__ segmax, int E, int Hh) {
  int e = blockIdx.x * 8 + (threadIdx.x >> 5);
  if (e >= E) return;
  int lane = threadIdx.x & 31;
  int s = src[e], d = dst[e];
  size_t F = (size_t)Hh * 64;
  for (int h = 0; h < Hh; ++h) {
    float acc = 0.f;
#pragma unroll
    for (int c = lane; c < 64; c += 32) {
      float v = xl[(size_t)s * F + h * 64 + c] + xr[(size_t)d * F + h * 64 + c];
      v = (v > 0.f) ? v : 0.2f * v;             // leaky_relu(0.2)
      acc += v * att[h * 64 + c];
    }
    acc += __shfl_down(acc, 16);
    acc += __shfl_down(acc, 8);
    acc += __shfl_down(acc, 4);
    acc += __shfl_down(acc, 2);
    acc += __shfl_down(acc, 1);
    if (lane == 0) {
      logit[(size_t)e * Hh + h] = acc;
      atomicMax(&segmax[(size_t)d * Hh + h], fkey(acc));
    }
  }
}

__global__ void edge_exp(float* __restrict__ logit, const int* __restrict__ dst,
                         const unsigned* __restrict__ segmax, float* __restrict__ segsum,
                         int E, int Hh) {
  long i = (long)blockIdx.x * blockDim.x + threadIdx.x;
  if (i >= (long)E * Hh) return;
  int e = (int)(i / Hh), h = (int)(i - (long)e * Hh);
  int d = dst[e];
  float m = funkey(segmax[(size_t)d * Hh + h]);
  float ex = __expf(logit[i] - m);
  logit[i] = ex;
  atomicAdd(&segsum[(size_t)d * Hh + h], ex);
}

__global__ void edge_scatter(const float* __restrict__ xl, const int* __restrict__ src,
                             const int* __restrict__ dst, const float* __restrict__ exv,
                             const float* __restrict__ segsum, float* __restrict__ outb,
                             int E, int Hh) {
  int e = blockIdx.x * 8 + (threadIdx.x >> 5);
  if (e >= E) return;
  int lane = threadIdx.x & 31;
  int s = src[e], d = dst[e];
  size_t F = (size_t)Hh * 64;
  for (int h = 0; h < Hh; ++h) {
    float a = exv[(size_t)e * Hh + h] / segsum[(size_t)d * Hh + h];
#pragma unroll
    for (int c = lane; c < 64; c += 32)
      atomicAdd(&outb[(size_t)d * F + h * 64 + c], a * xl[(size_t)s * F + h * 64 + c]);
  }
}

// ------------------------------------------------------- LSTM cell + head
__global__ void lstm_cell(const float* __restrict__ gates, const float* __restrict__ bih,
                          const float* __restrict__ bhh, float* __restrict__ h,
                          float* __restrict__ cst, int n) {
  long i = (long)blockIdx.x * blockDim.x + threadIdx.x;
  if (i >= (long)n * 64) return;
  int node = (int)(i >> 6), j = (int)(i & 63);
  const float* g = gates + (size_t)node * 256;
  float gi = g[j]        + bih[j]        + bhh[j];
  float gf = g[64 + j]   + bih[64 + j]   + bhh[64 + j];
  float gc = g[128 + j]  + bih[128 + j]  + bhh[128 + j];
  float go = g[192 + j]  + bih[192 + j]  + bhh[192 + j];
  float cn = sigm(gf) * cst[i] + sigm(gi) * tanhf(gc);
  cst[i] = cn;
  h[i] = sigm(go) * tanhf(cn);
}

__global__ void head_out(const float* __restrict__ fc1, const float* __restrict__ Wout,
                         const float* __restrict__ bout, float* __restrict__ out, int n) {
  int i = blockIdx.x * blockDim.x + threadIdx.x;
  if (i >= n) return;
  float acc = bout[0];
#pragma unroll
  for (int k = 0; k < 32; ++k) acc += fc1[(size_t)i * 32 + k] * Wout[k];
  out[i] = acc;
}

// ------------------------------------------------------------ launcher
extern "C" void kernel_launch(void* const* d_in, const int* in_sizes, int n_in,
                              void* d_out, int out_size, void* d_ws, size_t ws_size,
                              hipStream_t stream) {
  const float* x    = (const float*)d_in[0];
  const int*   ei   = (const int*)d_in[1];
  const float* Wl1  = (const float*)d_in[2];
  const float* Wr1  = (const float*)d_in[3];
  const float* att1 = (const float*)d_in[4];
  const float* b1   = (const float*)d_in[5];
  const float* Wl2  = (const float*)d_in[6];
  const float* Wr2  = (const float*)d_in[7];
  const float* att2 = (const float*)d_in[8];
  const float* b2   = (const float*)d_in[9];
  const float* Wih  = (const float*)d_in[10];
  const float* Whh  = (const float*)d_in[11];
  const float* bih  = (const float*)d_in[12];
  const float* bhh  = (const float*)d_in[13];
  const float* Wfc1 = (const float*)d_in[14];
  const float* bfc1 = (const float*)d_in[15];
  const float* Wout = (const float*)d_in[16];
  const float* bout = (const float*)d_in[17];

  const int N = in_sizes[0] / (TSEQ * FIN);   // 20000 (multiple of 16)
  const int E = in_sizes[1] / 2;              // 320000
  const int* src = ei;
  const int* dst = ei + E;

  // -------- workspace carve-up (~129 MB of f32-aligned slabs) --------
  float* ws = (float*)d_ws;
  size_t off = 0;
  auto allocf = [&](size_t n) { float* p = ws + off; off += (n + 63) & ~(size_t)63; return p; };
  float*    xl1    = allocf((size_t)N * 256);
  float*    xr1    = allocf((size_t)N * 256);
  float*    agg1   = allocf((size_t)N * 256);
  float*    xl2    = allocf((size_t)N * 64);
  float*    xr2    = allocf((size_t)N * 64);
  float*    seq    = allocf((size_t)N * 64 * TSEQ);
  float*    logitb = allocf((size_t)E * 4);
  unsigned* segmax = (unsigned*)allocf((size_t)N * 4);
  float*    segsum = allocf((size_t)N * 4);
  float*    hbuf   = allocf((size_t)N * 64);
  float*    cbuf   = allocf((size_t)N * 64);
  __bf16*   abf    = (__bf16*)allocf((size_t)N * 128);   // N*256 bf16 staging
  __bf16*   wb     = (__bf16*)allocf(60000);             // 120000 bf16 weight slots
  float*    gates  = xl1;   // free during LSTM phase
  float*    fc1b   = xr2;   // free during head phase

  __bf16* Wl1t  = wb;                 // 256x64
  __bf16* Wr1t  = wb + 16384;         // 256x64
  __bf16* Wl2t  = wb + 32768;         // 64x256
  __bf16* Wr2t  = wb + 49152;         // 64x256
  __bf16* WihB  = wb + 65536;         // 256x64 (Wih is already (Wih.T)^T)
  __bf16* WhhB  = wb + 81920;         // 256x64
  __bf16* Wfc1t = wb + 98304;         // 32x64

  const int TB = 256;
  // -------- weight conversion (once per launch, deterministic) --------
  transpose_cvt_bf16<<<CDIV(64 * 256, TB), TB, 0, stream>>>(Wl1, Wl1t, 64, 256);
  transpose_cvt_bf16<<<CDIV(64 * 256, TB), TB, 0, stream>>>(Wr1, Wr1t, 64, 256);
  transpose_cvt_bf16<<<CDIV(256 * 64, TB), TB, 0, stream>>>(Wl2, Wl2t, 256, 64);
  transpose_cvt_bf16<<<CDIV(256 * 64, TB), TB, 0, stream>>>(Wr2, Wr2t, 256, 64);
  cvt_rows_bf16<<<CDIV(256 * 64, TB), TB, 0, stream>>>(Wih, WihB, 256 * 64, 64, 64, 0);
  cvt_rows_bf16<<<CDIV(256 * 64, TB), TB, 0, stream>>>(Whh, WhhB, 256 * 64, 64, 64, 0);
  transpose_cvt_bf16<<<CDIV(64 * 32, TB), TB, 0, stream>>>(Wfc1, Wfc1t, 64, 32);

  dim3 g1(N / 16, 16);   // N x 256 tiles
  dim3 g2(N / 16, 4);    // N x 64 tiles
  dim3 g3(N / 16, 2);    // N x 32 tiles
  const int EW = CDIV(E, 8);   // wave-per-edge grids

  // -------- GAT encoder over T timesteps --------
  for (int t = 0; t < TSEQ; ++t) {
    // layer 1: x_t -> xl1/xr1 via WMMA
    cvt_rows_bf16<<<CDIV((long)N * 64, TB), TB, 0, stream>>>(x, abf, (long)N * 64, 64,
                                                             (long)TSEQ * 64, (long)t * 64);
    gemm_bf16<<<g1, 32, 0, stream>>>(abf, Wl1t, xl1, N, 64, 256, 0);
    gemm_bf16<<<g1, 32, 0, stream>>>(abf, Wr1t, xr1, N, 64, 256, 0);
    fill_u32<<<CDIV((long)N * 4, TB), TB, 0, stream>>>(segmax, 0u, (long)N * 4);
    fill_f32<<<CDIV((long)N * 4, TB), TB, 0, stream>>>(segsum, 0.f, (long)N * 4);
    edge_logit<<<EW, 256, 0, stream>>>(xl1, xr1, src, dst, att1, logitb, segmax, E, 4);
    edge_exp<<<CDIV((long)E * 4, TB), TB, 0, stream>>>(logitb, dst, segmax, segsum, E, 4);
    fill_f32<<<CDIV((long)N * 256, TB), TB, 0, stream>>>(agg1, 0.f, (long)N * 256);
    edge_scatter<<<EW, 256, 0, stream>>>(xl1, src, dst, logitb, segsum, agg1, E, 4);
    bias_act<<<CDIV((long)N * 256, TB), TB, 0, stream>>>(agg1, b1, (long)N * 256, 256, 0);

    // layer 2: h1 -> xl2/xr2 via WMMA (K=256)
    cvt_rows_bf16<<<CDIV((long)N * 256, TB), TB, 0, stream>>>(agg1, abf, (long)N * 256, 256, 256, 0);
    gemm_bf16<<<g2, 32, 0, stream>>>(abf, Wl2t, xl2, N, 256, 64, 0);
    gemm_bf16<<<g2, 32, 0, stream>>>(abf, Wr2t, xr2, N, 256, 64, 0);
    fill_u32<<<CDIV((long)N, TB), TB, 0, stream>>>(segmax, 0u, (long)N);
    fill_f32<<<CDIV((long)N, TB), TB, 0, stream>>>(segsum, 0.f, (long)N);
    edge_logit<<<EW, 256, 0, stream>>>(xl2, xr2, src, dst, att2, logitb, segmax, E, 1);
    edge_exp<<<CDIV((long)E, TB), TB, 0, stream>>>(logitb, dst, segmax, segsum, E, 1);
    float* st = seq + (size_t)t * N * 64;
    fill_f32<<<CDIV((long)N * 64, TB), TB, 0, stream>>>(st, 0.f, (long)N * 64);
    edge_scatter<<<EW, 256, 0, stream>>>(xl2, src, dst, logitb, segsum, st, E, 1);
    bias_act<<<CDIV((long)N * 64, TB), TB, 0, stream>>>(st, b2, (long)N * 64, 64, 0);
  }

  // -------- LSTM over T (WMMA gate GEMMs, elementwise cell) --------
  fill_f32<<<CDIV((long)N * 64, TB), TB, 0, stream>>>(hbuf, 0.f, (long)N * 64);
  fill_f32<<<CDIV((long)N * 64, TB), TB, 0, stream>>>(cbuf, 0.f, (long)N * 64);
  __bf16* abf2 = abf + (size_t)N * 64;
  for (int t = 0; t < TSEQ; ++t) {
    const float* st = seq + (size_t)t * N * 64;
    cvt_rows_bf16<<<CDIV((long)N * 64, TB), TB, 0, stream>>>(st, abf, (long)N * 64, 64, 64, 0);
    gemm_bf16<<<g1, 32, 0, stream>>>(abf, WihB, gates, N, 64, 256, 0);
    cvt_rows_bf16<<<CDIV((long)N * 64, TB), TB, 0, stream>>>(hbuf, abf2, (long)N * 64, 64, 64, 0);
    gemm_bf16<<<g1, 32, 0, stream>>>(abf2, WhhB, gates, N, 64, 256, 1);  // accumulate
    lstm_cell<<<CDIV((long)N * 64, TB), TB, 0, stream>>>(gates, bih, bhh, hbuf, cbuf, N);
  }

  // -------- MLP head --------
  cvt_rows_bf16<<<CDIV((long)N * 64, TB), TB, 0, stream>>>(hbuf, abf, (long)N * 64, 64, 64, 0);
  gemm_bf16<<<g3, 32, 0, stream>>>(abf, Wfc1t, fc1b, N, 64, 32, 0);
  bias_act<<<CDIV((long)N * 32, TB), TB, 0, stream>>>(fc1b, bfc1, (long)N * 32, 32, 1);
  head_out<<<CDIV(N, TB), TB, 0, stream>>>(fc1b, Wout, bout, (float*)d_out, N);
}
